// ConstituencyTreeLSTM_44324062495020
// MI455X (gfx1250) — matrix operand order repaired and checked by reading, not compiled, not run on hardware
//
#include <hip/hip_runtime.h>
#include <hip/hip_bf16.h>

typedef _Float16 half8  __attribute__((ext_vector_type(8)));
typedef _Float16 half16 __attribute__((ext_vector_type(16)));
typedef float    float8 __attribute__((ext_vector_type(8)));

#define WDIM 300
#define TDIM 50
#define MDIM 168

// ---- GEMM1 (leaf): K layout [e 0:300 | tag 300:350 | tagp 350:400 | h_prev 400:568 | pad:576]
//      cols: 4*M = 672 padded to 768 so all 8 waves get exactly 6 tiles (no guards).
#define K1      576
#define C1COLS  672
#define C1P     768
#define K1T     (K1 / 32)      // 18
#define U1      (C1P / 16 / 8) // 6 tiles per wave, exact

// ---- GEMM2 (node): K layout [tag 0:50 | tagp 50:100 | h1 100:268 | k 268:436 | pad:448]
//      cols: 5*M = 840 padded to 896 so all 8 waves get exactly 7 tiles (no guards).
#define K2      448
#define C2COLS  840
#define C2P     896
#define K2T     (K2 / 32)      // 14
#define U2      (C2P / 16 / 8) // 7 tiles per wave, exact

#define ROWS    16
#define NWAVES  8
#define BLOCK   256

#define W1T_BYTES (C1P * K1 * 2)   // 884736
#define W2T_BYTES (C2P * K2 * 2)   // 802816

__device__ __forceinline__ float sigf(float x) { return 1.0f / (1.0f + __expf(-x)); }

// Pack/transpose/pad all weights into f16 column-major (B-operand friendly) buffers in d_ws.
// Pad rows (K) and pad columns are written as 0 so they contribute nothing.
__global__ void prep_weights(const float* __restrict__ We_l, const float* __restrict__ Wt_l,
                             const float* __restrict__ Wtp_l, const float* __restrict__ Uh_l,
                             const float* __restrict__ Wt_n, const float* __restrict__ Wtp_n,
                             const float* __restrict__ Uh_n, const float* __restrict__ Uk_n,
                             _Float16* __restrict__ W1T, _Float16* __restrict__ W2T) {
  int idx = blockIdx.x * blockDim.x + threadIdx.x;
  const int n1 = C1P * K1;
  const int n2 = C2P * K2;
  if (idx < n1) {
    int col = idx / K1, kk = idx % K1;
    float v = 0.0f;
    if (col < C1COLS) {
      if (kk < WDIM)                      v = We_l [ kk               * C1COLS + col];
      else if (kk < WDIM + TDIM)          v = Wt_l [(kk - WDIM)       * C1COLS + col];
      else if (kk < WDIM + 2 * TDIM)      v = Wtp_l[(kk - WDIM - TDIM)* C1COLS + col];
      else if (kk < WDIM + 2*TDIM + MDIM) v = Uh_l [(kk - WDIM - 2*TDIM) * C1COLS + col];
    }
    W1T[col * K1 + kk] = (_Float16)v;
  } else if (idx < n1 + n2) {
    int j = idx - n1;
    int col = j / K2, kk = j % K2;
    float v = 0.0f;
    if (col < C2COLS) {
      if (kk < TDIM)                   v = Wt_n [ kk              * C2COLS + col];
      else if (kk < 2 * TDIM)          v = Wtp_n[(kk - TDIM)      * C2COLS + col];
      else if (kk < 2 * TDIM + MDIM)   v = Uh_n [(kk - 2 * TDIM)  * C2COLS + col];
      else if (kk < 2 * TDIM + 2*MDIM) { if (col < 4 * MDIM) v = Uk_n[(kk - 2*TDIM - MDIM) * (4 * MDIM) + col]; }
    }
    W2T[col * K2 + kk] = (_Float16)v;
  }
}

__global__ __launch_bounds__(BLOCK)
void treelstm_fused(const float* __restrict__ e, const float* __restrict__ tag,
                    const float* __restrict__ tagp, const float* __restrict__ h_prev,
                    const float* __restrict__ c_prev, const float* __restrict__ k_in,
                    const float* __restrict__ q, const float* __restrict__ b_l,
                    const float* __restrict__ b_n,
                    const _Float16* __restrict__ W1T, const _Float16* __restrict__ W2T,
                    float* __restrict__ out, int Ntot) {
  // LDS carve-out: A-panel (aliased between phases) | PRE (f32 pre-activations) | C1 (leaf cell state)
  __shared__ __align__(32) unsigned char smem[ROWS * K1 * 2 + ROWS * C2P * 4 + ROWS * MDIM * 4];
  _Float16* Abuf = (_Float16*)smem;                                  // 16x576 f16, later 16x448 f16
  float*    PRE  = (float*)(smem + ROWS * K1 * 2);                   // 16x896 f32
  float*    C1s  = (float*)(smem + ROWS * K1 * 2 + ROWS * C2P * 4);  // 16x168 f32

  const int tid     = threadIdx.x;
  const int lane    = tid & 31;
  // Uniform (SGPR) wave index: tile assignment stays scalar, EXEC untouched around WMMA.
  const int wv      = __builtin_amdgcn_readfirstlane(tid >> 5);
  const int rowBase = blockIdx.x * ROWS;

  // ===== Stage A-panel for GEMM1 (f16, K padded to 576) =====
  for (int j = tid; j < ROWS * K1; j += BLOCK) {
    int r = j / K1, c = j % K1;
    int gr = rowBase + r; if (gr >= Ntot) gr = Ntot - 1;
    float v = 0.0f;
    if (c < WDIM)                      v = e     [gr * WDIM + c];
    else if (c < WDIM + TDIM)          v = tag   [gr * TDIM + (c - WDIM)];
    else if (c < WDIM + 2 * TDIM)      v = tagp  [gr * TDIM + (c - WDIM - TDIM)];
    else if (c < WDIM + 2*TDIM + MDIM) v = h_prev[gr * MDIM + (c - WDIM - 2 * TDIM)];
    Abuf[r * K1 + c] = (_Float16)v;
  }
  __syncthreads();

  // Fragment addressing (wave32; this code's contiguous K order, same for A and B):
  //  A lane<16: row=lane,    K 0..15 contiguous;       lane>=16: row=lane-16, K 16..31 contiguous
  //  B lane<16: col=lane,    K 0..15 contiguous;       lane>=16: col=lane-16, K 16..31 contiguous
  //  C vgpr v:  M = v + 8*(lane>=16), N = lane&15
  const int aRow  = lane & 15;
  const int aOff  = (lane & 16) ? 16 : 0;
  const int bOff  = (lane & 16) ? 16 : 0;
  const int bCol0 = lane & 15;
  const int cN    = lane & 15;
  const int mBase = (lane & 16) ? 8 : 0;
  const float8 zero = {};

  // ===== GEMM1: pre[16 x 768] = A1[16 x 576] * W1 (software-pipelined, branch-free) =====
  {
    float8 acc[U1];
    #pragma unroll
    for (int u = 0; u < U1; ++u) acc[u] = zero;
    const _Float16* bBase = &W1T[(wv * 16 + bCol0) * K1 + bOff];
    const _Float16* aBase = &Abuf[aRow * K1 + aOff];

    half16 bcur[U1], acur;
    {
      half8 a0 = *(const half8*)aBase;
      half8 a1 = *(const half8*)(aBase + 8);
      acur = __builtin_shufflevector(a0, a1, 0,1,2,3,4,5,6,7,8,9,10,11,12,13,14,15);
      #pragma unroll
      for (int u = 0; u < U1; ++u)
        bcur[u] = *(const half16*)(bBase + u * (NWAVES * 16 * K1));
    }
    #pragma unroll
    for (int kt = 0; kt < K1T; ++kt) {
      half16 bnxt[U1], anxt;
      if (kt + 1 < K1T) {                      // resolved at compile time (full unroll)
        half8 a0 = *(const half8*)(aBase + (kt + 1) * 32);
        half8 a1 = *(const half8*)(aBase + (kt + 1) * 32 + 8);
        anxt = __builtin_shufflevector(a0, a1, 0,1,2,3,4,5,6,7,8,9,10,11,12,13,14,15);
        #pragma unroll
        for (int u = 0; u < U1; ++u)
          bnxt[u] = *(const half16*)(bBase + u * (NWAVES * 16 * K1) + (kt + 1) * 32);
      }
      #pragma unroll
      for (int u = 0; u < U1; ++u)
        acc[u] = __builtin_amdgcn_wmma_f32_16x16x32_f16(false, acur, false, bcur[u], (short)0, acc[u], false, false);
      if (kt + 1 < K1T) {
        acur = anxt;
        #pragma unroll
        for (int u = 0; u < U1; ++u) bcur[u] = bnxt[u];
      }
    }
    #pragma unroll
    for (int u = 0; u < U1; ++u) {
      int ct = wv + u * NWAVES;
      #pragma unroll
      for (int v = 0; v < 8; ++v)
        PRE[(mBase + v) * C2P + ct * 16 + cN] = acc[u][v];
    }
  }
  __syncthreads();

  // ===== Stage static part of GEMM2 A-panel (aliases GEMM1 A-panel; safe after barrier) =====
  for (int j = tid; j < ROWS * K2; j += BLOCK) {
    int r = j / K2, c = j % K2;
    int gr = rowBase + r; if (gr >= Ntot) gr = Ntot - 1;
    if (c >= 2 * TDIM && c < 2 * TDIM + MDIM) continue;   // h1 slot, filled below
    float v = 0.0f;
    if (c < TDIM)                    v = tag [gr * TDIM + c];
    else if (c < 2 * TDIM)           v = tagp[gr * TDIM + (c - TDIM)];
    else if (c < 2 * TDIM + 2*MDIM)  v = k_in[gr * MDIM + (c - 2 * TDIM - MDIM)];
    Abuf[r * K2 + c] = (_Float16)v;
  }

  // ===== Leaf gates: c1 = sig(i)*tanh(u) + sig(fl)*c_prev ; h1 = sig(o)*tanh(c1) =====
  for (int j = tid; j < ROWS * MDIM; j += BLOCK) {
    int r = j / MDIM, c = j % MDIM;
    int gr = rowBase + r; if (gr >= Ntot) gr = Ntot - 1;
    float iv = PRE[r * C2P +            c] + b_l[           c];
    float ov = PRE[r * C2P + MDIM     + c] + b_l[MDIM     + c];
    float fv = PRE[r * C2P + 2*MDIM   + c] + b_l[2*MDIM   + c];
    float uv = PRE[r * C2P + 3*MDIM   + c] + b_l[3*MDIM   + c];
    float c1 = sigf(iv) * tanhf(uv) + sigf(fv) * c_prev[gr * MDIM + c];
    float h1 = sigf(ov) * tanhf(c1);
    C1s[r * MDIM + c] = c1;
    Abuf[r * K2 + 2 * TDIM + c] = (_Float16)h1;
  }
  __syncthreads();

  // ===== GEMM2: pre5(+Uk term) [16 x 896] = A2[16 x 448] * W2 (software-pipelined) =====
  {
    float8 acc[U2];
    #pragma unroll
    for (int u = 0; u < U2; ++u) acc[u] = zero;
    const _Float16* bBase = &W2T[(wv * 16 + bCol0) * K2 + bOff];
    const _Float16* aBase = &Abuf[aRow * K2 + aOff];

    half16 bcur[U2], acur;
    {
      half8 a0 = *(const half8*)aBase;
      half8 a1 = *(const half8*)(aBase + 8);
      acur = __builtin_shufflevector(a0, a1, 0,1,2,3,4,5,6,7,8,9,10,11,12,13,14,15);
      #pragma unroll
      for (int u = 0; u < U2; ++u)
        bcur[u] = *(const half16*)(bBase + u * (NWAVES * 16 * K2));
    }
    #pragma unroll
    for (int kt = 0; kt < K2T; ++kt) {
      half16 bnxt[U2], anxt;
      if (kt + 1 < K2T) {
        half8 a0 = *(const half8*)(aBase + (kt + 1) * 32);
        half8 a1 = *(const half8*)(aBase + (kt + 1) * 32 + 8);
        anxt = __builtin_shufflevector(a0, a1, 0,1,2,3,4,5,6,7,8,9,10,11,12,13,14,15);
        #pragma unroll
        for (int u = 0; u < U2; ++u)
          bnxt[u] = *(const half16*)(bBase + u * (NWAVES * 16 * K2) + (kt + 1) * 32);
      }
      #pragma unroll
      for (int u = 0; u < U2; ++u)
        acc[u] = __builtin_amdgcn_wmma_f32_16x16x32_f16(false, acur, false, bcur[u], (short)0, acc[u], false, false);
      if (kt + 1 < K2T) {
        acur = anxt;
        #pragma unroll
        for (int u = 0; u < U2; ++u) bcur[u] = bnxt[u];
      }
    }
    #pragma unroll
    for (int u = 0; u < U2; ++u) {
      int ct = wv + u * NWAVES;
      #pragma unroll
      for (int v = 0; v < 8; ++v)
        PRE[(mBase + v) * C2P + ct * 16 + cN] = acc[u][v];
    }
  }
  __syncthreads();

  // ===== Node gates + output: c2 = sig(i2)tanh(u2) + sig(fd2)q + sig(fl2)c1 ; h2 = sig(o2)tanh(c2) =====
  for (int j = tid; j < ROWS * MDIM; j += BLOCK) {
    int r = j / MDIM, c = j % MDIM;
    int gr = rowBase + r;
    if (gr >= Ntot) continue;
    float i2  = PRE[r * C2P +            c] + b_n[           c];
    float o2  = PRE[r * C2P + MDIM     + c] + b_n[MDIM     + c];
    float fl2 = PRE[r * C2P + 2*MDIM   + c] + b_n[2*MDIM   + c];
    float fd2 = PRE[r * C2P + 3*MDIM   + c] + b_n[3*MDIM   + c];
    float u2  = PRE[r * C2P + 4*MDIM   + c] + b_n[4*MDIM   + c];
    float c2 = sigf(i2) * tanhf(u2) + sigf(fd2) * q[gr * MDIM + c] + sigf(fl2) * C1s[r * MDIM + c];
    float h2 = sigf(o2) * tanhf(c2);
    out[gr * (2 * MDIM) +        c] = h2;
    out[gr * (2 * MDIM) + MDIM + c] = c2;
  }
}

extern "C" void kernel_launch(void* const* d_in, const int* in_sizes, int n_in,
                              void* d_out, int out_size, void* d_ws, size_t ws_size,
                              hipStream_t stream) {
  const float* e      = (const float*)d_in[0];
  const float* tag    = (const float*)d_in[1];
  const float* tagp   = (const float*)d_in[2];
  const float* h_prev = (const float*)d_in[3];
  const float* c_prev = (const float*)d_in[4];
  const float* k_in   = (const float*)d_in[5];
  const float* q      = (const float*)d_in[6];
  const float* We_l   = (const float*)d_in[7];
  const float* Wt_l   = (const float*)d_in[8];
  const float* Wtp_l  = (const float*)d_in[9];
  const float* Uh_l   = (const float*)d_in[10];
  const float* b_l    = (const float*)d_in[11];
  const float* Wt_n   = (const float*)d_in[12];
  const float* Wtp_n  = (const float*)d_in[13];
  const float* Uh_n   = (const float*)d_in[14];
  const float* Uk_n   = (const float*)d_in[15];
  const float* b_n    = (const float*)d_in[16];

  _Float16* W1T = (_Float16*)d_ws;
  _Float16* W2T = (_Float16*)((char*)d_ws + W1T_BYTES);

  const int N = in_sizes[0] / WDIM;

  const int prepN = C1P * K1 + C2P * K2;
  prep_weights<<<(prepN + 255) / 256, 256, 0, stream>>>(
      We_l, Wt_l, Wtp_l, Uh_l, Wt_n, Wtp_n, Uh_n, Uk_n, W1T, W2T);

  const int nblocks = (N + ROWS - 1) / ROWS;   // 6250 for N=100000
  treelstm_fused<<<nblocks, BLOCK, 0, stream>>>(
      e, tag, tagp, h_prev, c_prev, k_in, q, b_l, b_n, W1T, W2T, (float*)d_out, N);
}
